// RibonanzaNetAttention_17325898072627
// MI455X (gfx1250) — matrix operand order repaired and checked by reading, not compiled, not run on hardware
//
#include <hip/hip_runtime.h>
#include <hip/hip_bf16.h>

#define BB 2
#define SS 2048
#define HN 12
#define HD 64
#define DM 768
#define MTOT (BB * SS)
#define SCALE 0.125f
#define LN_EPS 1e-12f

typedef __attribute__((ext_vector_type(16))) __bf16 bf16x16;
typedef __attribute__((ext_vector_type(2)))  __bf16 bf16x2;
typedef __attribute__((ext_vector_type(8)))  float  f32x8;

static __device__ __forceinline__ f32x8 wmma_bf16(bf16x16 a, bf16x16 b, f32x8 c) {
  return __builtin_amdgcn_wmma_f32_16x16x32_bf16(false, a, false, b, (short)0, c,
                                                 false, false);
}

// Async 16-byte global->LDS copy (ASYNCcnt-tracked).  LDS byte offset is the
// low 32 bits of the generic pointer (aperture bits live in the high word).
static __device__ __forceinline__ void async_ld16(unsigned lds_off, const void* g) {
  asm volatile("global_load_async_to_lds_b128 %0, %1, off"
               :: "v"(lds_off), "v"((unsigned long long)(uintptr_t)g)
               : "memory");
}
static __device__ __forceinline__ void wait_async0() {
  asm volatile("s_wait_asynccnt 0" ::: "memory");
}

// A-matrix fragment (16x32 bf16) per CDNA5 layout:
// lanes 0-15: M=lane, K in [0..7] then [16..23]; lanes 16-31: K in [8..15],[24..31]
template <typename T>
static __device__ __forceinline__ bf16x16 load_a16(const T* base, int ld, int koff) {
  const int lane = threadIdx.x & 31;
  const int m = lane & 15;
  const int half = lane >> 4;
  const T* p = base + (long)m * ld + koff + half * 8;
  bf16x16 a;
#pragma unroll
  for (int j = 0; j < 8; ++j) {
    a[j]     = (__bf16)p[j];
    a[8 + j] = (__bf16)p[16 + j];
  }
  return a;
}

// B-matrix fragment (32x16 bf16): column n = lane&15;
// lanes 0-15: K in [0..7] then [16..23]; lanes 16-31: K in [8..15],[24..31].
template <typename T>
static __device__ __forceinline__ bf16x16 load_b16(const T* base, long colStride,
                                                   int koff) {
  const int lane = threadIdx.x & 31;
  const int n = lane & 15;
  const int khalf = lane >> 4;
  const T* p = base + (long)n * colStride + koff + khalf * 8;
  bf16x16 b;
#pragma unroll
  for (int j = 0; j < 8; ++j) {
    b[j]     = (__bf16)p[j];
    b[8 + j] = (__bf16)p[16 + j];
  }
  return b;
}

// ---------------------------------------------------------------------------
// Kernel 1: fused QKV projection.  y = x @ W{q,k,v}, bf16 outputs.
// Q,K -> [B,H,S,HD]; V -> transposed [B,H,HD,S].
// grid (MTOT/128, DM/16, 3), block 256 (8 waves).
// ---------------------------------------------------------------------------
__global__ __launch_bounds__(256) void qkv_proj_kernel(
    const float* __restrict__ x, const float* __restrict__ Wq,
    const float* __restrict__ Wk, const float* __restrict__ Wv,
    __bf16* __restrict__ qo, __bf16* __restrict__ ko, __bf16* __restrict__ vTo) {
  __shared__ __bf16 wtile[16 * 34];  // [n][k] transposed, even pitch for b32 stores
  const int wave = threadIdx.x >> 5;
  const int lane = threadIdx.x & 31;
  const int n = lane & 15;
  const int half = lane >> 4;
  const int row0 = blockIdx.x * 128 + wave * 16;
  const int n0 = blockIdx.y * 16;
  const int which = blockIdx.z;
  const float* W = (which == 0) ? Wq : ((which == 1) ? Wk : Wv);
  const float* abase = x + (long)row0 * DM;

  const int snn = threadIdx.x & 15;         // staging: column
  const int skk = (threadIdx.x >> 4) << 1;  // staging: k-pair

  f32x8 acc = {};
  for (int k0 = 0; k0 < DM; k0 += 32) {
    __syncthreads();
    {
      float w0 = W[(long)(k0 + skk) * DM + n0 + snn];
      float w1 = W[(long)(k0 + skk + 1) * DM + n0 + snn];
      bf16x2 pk;
      pk[0] = (__bf16)w0;
      pk[1] = (__bf16)w1;
      *(bf16x2*)&wtile[snn * 34 + skk] = pk;
    }
    __syncthreads();
    acc = wmma_bf16(load_a16(abase, DM, k0),
                    load_b16((const __bf16*)wtile, 34, 0), acc);
  }
  const int c = n0 + n;
  const int hh = c / HD;
  const int dd = c % HD;
#pragma unroll
  for (int r = 0; r < 8; ++r) {
    int gm = row0 + half * 8 + r;
    int bb = gm >> 11;          // / SS
    int ss = gm & (SS - 1);
    __bf16 val = (__bf16)acc[r];
    if (which == 0)
      qo[(((long)bb * HN + hh) * SS + ss) * HD + dd] = val;
    else if (which == 1)
      ko[(((long)bb * HN + hh) * SS + ss) * HD + dd] = val;
    else
      vTo[(((long)bb * HN + hh) * HD + dd) * (long)SS + ss] = val;
  }
}

// ---------------------------------------------------------------------------
// Kernel 2: flash attention.  One block = (b, h, 128 q rows); its 8 waves own
// 8 consecutive 16-row q-tiles and SHARE the per-iteration 32-key K/V tiles,
// staged in LDS via global_load_async_to_lds_b128 (8x less L2 traffic).
// Per 32 keys: 4 WMMA (QK^T) + online softmax + 4 WMMA (PV).
// grid B*H*(S/128) = 384, block 256.
// ---------------------------------------------------------------------------
#define KP 72  // ktile pitch in bf16 (32 rows x 64 d + pad)
#define VP 40  // vtile pitch in bf16 (64 rows x 32 k + pad)

__global__ __launch_bounds__(256) void flash_attn_kernel(
    const __bf16* __restrict__ qi, const __bf16* __restrict__ ki,
    const __bf16* __restrict__ vT, const float* __restrict__ bias,
    const float* __restrict__ mask, float* __restrict__ ctx) {
  __shared__ __bf16 ktile[32 * KP];       // [key][d]
  __shared__ __bf16 vtile[64 * VP];       // [d][key]
  __shared__ __bf16 pstage[8][16 * 32];   // per-wave probs transpose tile

  const int tid = threadIdx.x;
  const int wave = tid >> 5;
  const int lane = tid & 31;
  const int n = lane & 15;
  const int half = lane >> 4;

  const int qblk = blockIdx.x & 15;            // S/128
  const int h = (blockIdx.x >> 4) % HN;
  const int b = blockIdx.x / (16 * HN);
  const int q0 = qblk * 128 + wave * 16;

  const __bf16* qbase = qi + ((((long)b * HN + h) * SS) + q0) * HD;
  const __bf16* kbase = ki + (((long)b * HN + h) * SS) * HD;
  const __bf16* vbase = vT + (((long)b * HN + h) * HD) * (long)SS;
  const float* biasbase = bias + ((((long)b * HN + h) * SS) + q0) * (long)SS;
  const float* maskbase = mask + (long)b * SS;

  // async staging assignments (16B per thread per tile)
  const int kkr = tid >> 3, chk = tid & 7;     // K: 32 rows x 8 chunks
  const int dvr = tid >> 2, chv = tid & 3;     // V: 64 rows x 4 chunks
  const unsigned kt_off = (unsigned)(uintptr_t)ktile + kkr * (KP * 2) + chk * 16;
  const unsigned vt_off = (unsigned)(uintptr_t)vtile + dvr * (VP * 2) + chv * 16;

  bf16x16 aq0 = load_a16(qbase, HD, 0);
  bf16x16 aq1 = load_a16(qbase, HD, 32);

  float row_max[8], row_sum[8];
#pragma unroll
  for (int r = 0; r < 8; ++r) { row_max[r] = -3.0e38f; row_sum[r] = 0.0f; }
  f32x8 acc[4] = {};

  __bf16* ps = &pstage[wave][0];

  for (int kb = 0; kb < SS; kb += 32) {
    __syncthreads();  // all waves done reading previous K/V tiles
    async_ld16(kt_off, kbase + (long)(kb + kkr) * HD + chk * 8);
    async_ld16(vt_off, vbase + (long)dvr * SS + kb + chv * 8);
    wait_async0();
    __syncthreads();  // tiles visible to all waves

    __builtin_prefetch(biasbase + (long)(half * 8) * SS + kb + 32, 0, 0);
    float sc[2][8];
#pragma unroll
    for (int t = 0; t < 2; ++t) {
      f32x8 s = {};
      s = wmma_bf16(aq0, load_b16((const __bf16*)&ktile[t * 16 * KP], KP, 0), s);
      s = wmma_bf16(aq1, load_b16((const __bf16*)&ktile[t * 16 * KP], KP, 32), s);
      const int kcol = kb + t * 16 + n;
      const float mk = maskbase[kcol];
#pragma unroll
      for (int r = 0; r < 8; ++r)
        sc[t][r] = s[r] * SCALE + biasbase[(long)(half * 8 + r) * SS + kcol] + mk;
    }
    // online softmax: row stats replicated across each 16-lane half
    float mx[8];
#pragma unroll
    for (int r = 0; r < 8; ++r) mx[r] = fmaxf(sc[0][r], sc[1][r]);
#pragma unroll
    for (int d = 1; d < 16; d <<= 1)
#pragma unroll
      for (int r = 0; r < 8; ++r) mx[r] = fmaxf(mx[r], __shfl_xor(mx[r], d, 32));
    float alpha[8], psum[8];
#pragma unroll
    for (int r = 0; r < 8; ++r) {
      float nm = fmaxf(row_max[r], mx[r]);
      alpha[r] = __expf(row_max[r] - nm);
      row_max[r] = nm;
      psum[r] = 0.0f;
    }
#pragma unroll
    for (int t = 0; t < 2; ++t)
#pragma unroll
      for (int r = 0; r < 8; ++r) {
        float p = __expf(sc[t][r] - row_max[r]);
        sc[t][r] = p;
        psum[r] += p;
      }
#pragma unroll
    for (int d = 1; d < 16; d <<= 1)
#pragma unroll
      for (int r = 0; r < 8; ++r) psum[r] += __shfl_xor(psum[r], d, 32);
#pragma unroll
    for (int r = 0; r < 8; ++r) row_sum[r] = row_sum[r] * alpha[r] + psum[r];
#pragma unroll
    for (int dt = 0; dt < 4; ++dt)
#pragma unroll
      for (int r = 0; r < 8; ++r) acc[dt][r] *= alpha[r];
    // stage probs (C-layout) -> 16x32 bf16 row-major tile (per-wave region)
#pragma unroll
    for (int t = 0; t < 2; ++t)
#pragma unroll
      for (int r = 0; r < 8; ++r)
        ps[(half * 8 + r) * 32 + t * 16 + n] = (__bf16)sc[t][r];
    asm volatile("s_wait_dscnt 0" ::: "memory");  // wave-local LDS RAW fence
    bf16x16 ap = load_a16((const __bf16*)ps, 32, 0);
#pragma unroll
    for (int dt = 0; dt < 4; ++dt)
      acc[dt] = wmma_bf16(ap, load_b16((const __bf16*)&vtile[dt * 16 * VP], VP, 0),
                          acc[dt]);
  }
  // epilogue: normalize, store ctx as [B,S,H,HD] (flattens to [M, DM])
#pragma unroll
  for (int dt = 0; dt < 4; ++dt)
#pragma unroll
    for (int r = 0; r < 8; ++r) {
      int row = half * 8 + r;
      float val = acc[dt][r] / row_sum[r];
      ctx[((((long)b * SS) + q0 + row) * HN + h) * HD + dt * 16 + n] = val;
    }
}

// ---------------------------------------------------------------------------
// Kernel 3: output projection + first residual.  t = ctx @ Wo + x (fp32).
// grid (MTOT/128, DM/16), block 256.
// ---------------------------------------------------------------------------
__global__ __launch_bounds__(256) void out_proj_kernel(
    const float* __restrict__ ctx, const float* __restrict__ Wo,
    const float* __restrict__ x, float* tout) {
  __shared__ __bf16 wtile[16 * 34];
  const int wave = threadIdx.x >> 5;
  const int lane = threadIdx.x & 31;
  const int n = lane & 15;
  const int half = lane >> 4;
  const int row0 = blockIdx.x * 128 + wave * 16;
  const int n0 = blockIdx.y * 16;
  const float* abase = ctx + (long)row0 * DM;
  const int snn = threadIdx.x & 15;
  const int skk = (threadIdx.x >> 4) << 1;

  f32x8 acc = {};
  for (int k0 = 0; k0 < DM; k0 += 32) {
    __syncthreads();
    {
      float w0 = Wo[(long)(k0 + skk) * DM + n0 + snn];
      float w1 = Wo[(long)(k0 + skk + 1) * DM + n0 + snn];
      bf16x2 pk;
      pk[0] = (__bf16)w0;
      pk[1] = (__bf16)w1;
      *(bf16x2*)&wtile[snn * 34 + skk] = pk;
    }
    __syncthreads();
    acc = wmma_bf16(load_a16(abase, DM, k0),
                    load_b16((const __bf16*)wtile, 34, 0), acc);
  }
  const int c = n0 + n;
#pragma unroll
  for (int r = 0; r < 8; ++r) {
    long gm = row0 + half * 8 + r;
    tout[gm * DM + c] = acc[r] + x[gm * DM + c];
  }
}

// ---------------------------------------------------------------------------
// Kernel 4: fused dual LayerNorm: out = LN2(LN1(t) + x).  Block per row;
// safe in-place on t==out (all reads precede writes within a row).
// ---------------------------------------------------------------------------
static __device__ __forceinline__ float block_sum256(float v, float* red) {
  const int tid = threadIdx.x;
  __syncthreads();
  red[tid] = v;
  __syncthreads();
  for (int off = 128; off > 0; off >>= 1) {
    if (tid < off) red[tid] += red[tid + off];
    __syncthreads();
  }
  return red[0];
}

__global__ __launch_bounds__(256) void dual_layernorm_kernel(
    const float* t, const float* __restrict__ x, const float* __restrict__ g1,
    const float* __restrict__ b1, const float* __restrict__ g2,
    const float* __restrict__ b2, float* out) {
  __shared__ float red[256];
  const int row = blockIdx.x;
  const int tid = threadIdx.x;
  const float* tr = t + (long)row * DM;
  const float* xr = x + (long)row * DM;
  float v[3], xv[3];
#pragma unroll
  for (int i = 0; i < 3; ++i) {
    v[i] = tr[tid + i * 256];
    xv[i] = xr[tid + i * 256];
  }
  float mean = block_sum256(v[0] + v[1] + v[2], red) * (1.0f / DM);
  float sq = 0.0f;
#pragma unroll
  for (int i = 0; i < 3; ++i) { v[i] -= mean; sq += v[i] * v[i]; }
  float inv1 = rsqrtf(block_sum256(sq, red) * (1.0f / DM) + LN_EPS);
  float z[3];
#pragma unroll
  for (int i = 0; i < 3; ++i) {
    int col = tid + i * 256;
    z[i] = v[i] * inv1 * g1[col] + b1[col] + xv[i];
  }
  float mean2 = block_sum256(z[0] + z[1] + z[2], red) * (1.0f / DM);
  float sq2 = 0.0f;
#pragma unroll
  for (int i = 0; i < 3; ++i) { z[i] -= mean2; sq2 += z[i] * z[i]; }
  float inv2 = rsqrtf(block_sum256(sq2, red) * (1.0f / DM) + LN_EPS);
#pragma unroll
  for (int i = 0; i < 3; ++i) {
    int col = tid + i * 256;
    out[(long)row * DM + col] = z[i] * inv2 * g2[col] + b2[col];
  }
}

// ---------------------------------------------------------------------------
extern "C" void kernel_launch(void* const* d_in, const int* in_sizes, int n_in,
                              void* d_out, int out_size, void* d_ws,
                              size_t ws_size, hipStream_t stream) {
  (void)in_sizes; (void)n_in; (void)out_size; (void)ws_size;
  const float* x    = (const float*)d_in[0];
  const float* mask = (const float*)d_in[1];
  const float* bias = (const float*)d_in[2];
  const float* Wq   = (const float*)d_in[3];
  const float* Wk   = (const float*)d_in[4];
  const float* Wv   = (const float*)d_in[5];
  const float* Wo   = (const float*)d_in[6];
  const float* g1   = (const float*)d_in[7];
  const float* b1   = (const float*)d_in[8];
  const float* g2   = (const float*)d_in[9];
  const float* b2   = (const float*)d_in[10];
  float* out = (float*)d_out;

  char* ws = (char*)d_ws;
  const size_t QKV_BYTES = (size_t)MTOT * DM * sizeof(__bf16);  // 6.29 MB each
  __bf16* qb  = (__bf16*)(ws);
  __bf16* kb  = (__bf16*)(ws + QKV_BYTES);
  __bf16* vTb = (__bf16*)(ws + 2 * QKV_BYTES);
  float*  ctx = (float*)(ws + 3 * QKV_BYTES);                   // 12.6 MB

  qkv_proj_kernel<<<dim3(MTOT / 128, DM / 16, 3), 256, 0, stream>>>(
      x, Wq, Wk, Wv, qb, kb, vTb);
  flash_attn_kernel<<<dim3(BB * HN * (SS / 128)), 256, 0, stream>>>(
      qb, kb, vTb, bias, mask, ctx);
  // d_out doubles as scratch for t = ctx @ Wo + x; LN then runs in-place
  out_proj_kernel<<<dim3(MTOT / 128, DM / 16), 256, 0, stream>>>(ctx, Wo, x, out);
  dual_layernorm_kernel<<<dim3(MTOT), 256, 0, stream>>>(out, x, g1, b1, g2, b2,
                                                        out);
}
